// ClEncoder_52364241273012
// MI455X (gfx1250) — matrix operand order repaired and checked by reading, not compiled, not run on hardware
//
#include <hip/hip_runtime.h>
#include <hip/hip_bf16.h>

typedef __attribute__((ext_vector_type(16))) _Float16 v16h;
typedef __attribute__((ext_vector_type(8)))  _Float16 v8h;
typedef __attribute__((ext_vector_type(8)))  float    v8f;
typedef __attribute__((ext_vector_type(4)))  float    v4f;

#define FEAT_DIM 2048
#define DL       64
#define HID      256            // 4*DIM_LATENT
#define NUM_USER 50000
#define NUM_ITEM 40000
#define NROWS    (NUM_USER + NUM_ITEM)
#define G_GROUPS 64
#define M_ROWS   512

#define KT1      (FEAT_DIM / 32)   // 64 K-tiles in GEMM1
#define NT1      (HID / 16)        // 16 N-tiles in GEMM1
#define TILE_H   (NT1 * 32 * 16)   // halves per B k-tile = 8192 (16 KB)
#define TILE_B   (TILE_H * 2)      // bytes per B k-tile   = 16384

// ---------------------------------------------------------------------------
// WMMA helper: D = A(16x32 f16) * B(32x16 f16) + C(16x16 f32)
// ---------------------------------------------------------------------------
__device__ __forceinline__ v8f wmma_f16(v16h a, v16h b, v8f c) {
  return __builtin_amdgcn_wmma_f32_16x16x32_f16(
      /*neg_a=*/false, a, /*neg_b=*/false, b,
      /*c_mod=*/(short)0, c, /*reuse_a=*/false, /*reuse_b=*/false);
}

// A-fragment from fp32 row-major source, converting to f16 in registers.
// ISA layout (16-bit A 16x32): lane L holds row m=L%16; element e (0..15):
//   k = (e/8)*16 + g*8 + (e%8),  g = L/16   -> two contiguous 8-float chunks.
__device__ __forceinline__ v16h load_a_f32(const float* __restrict__ row,
                                           int kbase, int g) {
  const v4f* p = (const v4f*)(row + kbase + g * 8);
  const v4f* q = (const v4f*)(row + kbase + g * 8 + 16);
  v4f f0 = p[0], f1 = p[1], f2 = q[0], f3 = q[1];
  v16h a;
#pragma unroll
  for (int i = 0; i < 4; ++i) {
    a[i]      = (_Float16)f0[i];
    a[i + 4]  = (_Float16)f1[i];
    a[i + 8]  = (_Float16)f2[i];
    a[i + 12] = (_Float16)f3[i];
  }
  return a;
}

// A-fragment from f16 row-major source (16B-aligned chunks).
__device__ __forceinline__ v16h load_a_f16(const _Float16* __restrict__ row,
                                           int kbase, int g) {
  v8h h0 = *(const v8h*)(row + kbase + g * 8);
  v8h h1 = *(const v8h*)(row + kbase + g * 8 + 16);
  v16h a;
#pragma unroll
  for (int i = 0; i < 8; ++i) { a[i] = h0[i]; a[i + 8] = h1[i]; }
  return a;
}

// ---------------------------------------------------------------------------
// Async copy of one 16 KB B k-tile into LDS.  256 threads x 4 x 16B.
// ISA: copies global VADDR+offset to LDS VDST+offset; completion counted by ASYNCcnt.
// ---------------------------------------------------------------------------
__device__ __forceinline__ void async_tile_load(const char* gsrc, void* ldst) {
  unsigned lds = (unsigned)(size_t)ldst;   // low 32 bits of a shared ptr = LDS offset
  asm volatile(
      "global_load_async_to_lds_b128 %0, %1, off\n\t"
      "global_load_async_to_lds_b128 %0, %1, off offset:4096\n\t"
      "global_load_async_to_lds_b128 %0, %1, off offset:8192\n\t"
      "global_load_async_to_lds_b128 %0, %1, off offset:12288"
      :: "v"(lds), "v"(gsrc) : "memory");
}

__device__ __forceinline__ void wait_async_all() {
  asm volatile("s_wait_asynccnt 0x0" ::: "memory");
}

// ---------------------------------------------------------------------------
// Pack a fp32 weight matrix W[K][N] into per-lane WMMA B-fragments (f16).
// ISA layout (16-bit B 32x16): lane L holds col n=L%16; element e: k=(e/8)*16+g*8+(e%8).
// ---------------------------------------------------------------------------
__global__ __launch_bounds__(256) void pack_w_kernel(
    const float* __restrict__ W, _Float16* __restrict__ out, int K, int N) {
  int t = blockIdx.x * blockDim.x + threadIdx.x;
  int NT = N >> 4;
  int total = (K >> 5) * NT * 32;
  if (t >= total) return;
  int lane = t & 31;
  int nt = (t >> 5) % NT;
  int kt = (t >> 5) / NT;
  int n = lane & 15, g = lane >> 4;
  _Float16* o = out + (size_t)t * 16;
#pragma unroll
  for (int e = 0; e < 16; ++e) {
    int k = kt * 32 + (e >> 3) * 16 + g * 8 + (e & 7);
    o[e] = (_Float16)W[(size_t)k * N + nt * 16 + n];
  }
}

// ---------------------------------------------------------------------------
// GEMM1: h1 = relu(features[40000x2048] @ W1 + b1) -> f16 [40000x256]
// 8 waves/block, 16 rows each (block = 128 rows x full N=256 in VGPRs).
// B k-tiles double-buffered in LDS via async-to-LDS copies; A streamed from
// HBM with inline fp32->f16 conversion.  Software-pipelined:
//   - A fragment for kt+1 loaded during kt's WMMAs (covers HBM latency)
//   - B fragments 2-deep from LDS (covers DS latency)
// ---------------------------------------------------------------------------
__global__ __launch_bounds__(256) void gemm1_kernel(
    const float* __restrict__ A, const _Float16* __restrict__ Bp,
    const float* __restrict__ bias, _Float16* __restrict__ O) {
  __shared__ __align__(16) _Float16 bbuf[2][TILE_H];

  const int tid  = threadIdx.x;
  const int lane = tid & 31;
  const int wave = tid >> 5;
  const int tile = blockIdx.x * 8 + wave;         // wave-uniform
  const bool active = (tile * 16 < NUM_ITEM);     // tail waves still barrier
  const int m0 = tile * 16;
  const int n = lane & 15, g = lane >> 4;
  const float* arow = A + (size_t)(m0 + n) * FEAT_DIM;

  v8f acc[NT1];
#pragma unroll
  for (int nt = 0; nt < NT1; ++nt) acc[nt] = (v8f){};

  // Prologue: stage B tile 0, preload A fragment 0.
  const char* bsrc = (const char*)Bp + tid * 16;
  async_tile_load(bsrc, (char*)&bbuf[0][0] + tid * 16);
  v16h a_cur = {}, a_nxt = {};
  if (active) a_cur = load_a_f32(arow, 0, g);
  wait_async_all();
  __syncthreads();

  for (int kt = 0; kt < KT1; ++kt) {
    const int cur = kt & 1;
    if (kt + 1 < KT1)   // prefetch next B k-tile into the other buffer
      async_tile_load(bsrc + (size_t)(kt + 1) * TILE_B,
                      (char*)&bbuf[cur ^ 1][0] + tid * 16);
    if (active) {
      if (kt + 1 < KT1)   // A fragment prefetch: hidden behind 16 WMMAs
        a_nxt = load_a_f32(arow, (kt + 1) * 32, g);
      const v16h* bp = ((const v16h*)&bbuf[cur][0]) + lane;
      v16h b0 = bp[0];
      v16h b1 = bp[32];
#pragma unroll
      for (int nt = 0; nt < NT1; ++nt) {
        v16h b2 = (nt + 2 < NT1) ? bp[(nt + 2) * 32] : b0;  // 2-deep DS pipe
        acc[nt] = wmma_f16(a_cur, b0, acc[nt]);
        b0 = b1;
        b1 = b2;
      }
      a_cur = a_nxt;
    }
    wait_async_all();    // own async loads done before signaling
    __syncthreads();     // whole tile valid + ping-pong buffer safe to reuse
  }

  if (!active) return;
  // C layout: lane L, VGPR r -> row m0 + (L/16)*8 + r, col nt*16 + (L%16)
  const int mbase = m0 + (lane >> 4) * 8;
#pragma unroll
  for (int nt = 0; nt < NT1; ++nt) {
    const float bz = bias[nt * 16 + n];
#pragma unroll
    for (int r = 0; r < 8; ++r) {
      float v = acc[nt][r] + bz;
      v = v > 0.0f ? v : 0.0f;
      O[(size_t)(mbase + r) * HID + nt * 16 + n] = (_Float16)v;
    }
  }
}

// ---------------------------------------------------------------------------
// GEMM2: h2 = relu(h1[40000x256] @ W2 + b2) -> f16 [40000x64]
// Flattened 8x4 (kt,nt) loop, B fragments 2-deep pipelined from L2.
// ---------------------------------------------------------------------------
__global__ __launch_bounds__(256) void gemm2_kernel(
    const _Float16* __restrict__ A, const _Float16* __restrict__ Bp,
    const float* __restrict__ bias, _Float16* __restrict__ O) {
  const int lane = threadIdx.x & 31;
  const int wave = threadIdx.x >> 5;
  const int tile = blockIdx.x * 8 + wave;
  if (tile * 16 >= NUM_ITEM) return;
  const int m0 = tile * 16;
  const int n = lane & 15, g = lane >> 4;
  const _Float16* arow = A + (size_t)(m0 + n) * HID;
  const v16h* bp = ((const v16h*)Bp) + lane;

  v8f acc[4];
#pragma unroll
  for (int nt = 0; nt < 4; ++nt) acc[nt] = (v8f){};

  v16h a_cur = load_a_f16(arow, 0, g), a_nxt = a_cur;
  v16h b0 = bp[0];
  v16h b1 = bp[32];
#pragma unroll
  for (int kt = 0; kt < HID / 32; ++kt) {
    if (kt + 1 < HID / 32) a_nxt = load_a_f16(arow, (kt + 1) * 32, g);
#pragma unroll
    for (int nt = 0; nt < 4; ++nt) {
      const int nxt = kt * 4 + nt + 2;
      v16h b2 = (nxt < (HID / 32) * 4) ? bp[nxt * 32] : b0;
      acc[nt] = wmma_f16(a_cur, b0, acc[nt]);
      b0 = b1;
      b1 = b2;
    }
    a_cur = a_nxt;
  }

  const int mbase = m0 + (lane >> 4) * 8;
#pragma unroll
  for (int nt = 0; nt < 4; ++nt) {
    const float bz = bias[nt * 16 + n];
#pragma unroll
    for (int r = 0; r < 8; ++r) {
      float v = acc[nt][r] + bz;
      v = v > 0.0f ? v : 0.0f;
      O[(size_t)(mbase + r) * DL + nt * 16 + n] = (_Float16)v;
    }
  }
}

// ---------------------------------------------------------------------------
// GEMM3: x1 = concat(users[50000x64], h2[40000x64]) @ W3 + b3 -> fp32 d_out
// 50000 and 40000 are multiples of 16 -> the source split is wave-uniform.
// ---------------------------------------------------------------------------
__global__ __launch_bounds__(256) void gemm3_kernel(
    const float* __restrict__ U, const _Float16* __restrict__ H2,
    const _Float16* __restrict__ Bp, const float* __restrict__ bias,
    float* __restrict__ O) {
  const int lane = threadIdx.x & 31;
  const int wave = threadIdx.x >> 5;
  const int tile = blockIdx.x * 8 + wave;
  if (tile >= NROWS / 16) return;
  const int m0 = tile * 16;
  const int n = lane & 15, g = lane >> 4;
  const v16h* bp = ((const v16h*)Bp) + lane;

  v8f acc[4];
#pragma unroll
  for (int nt = 0; nt < 4; ++nt) acc[nt] = (v8f){};

  v16h b0 = bp[0];
  v16h b1 = bp[32];
  if (m0 < NUM_USER) {
    const float* arow = U + (size_t)(m0 + n) * DL;
    v16h a_cur = load_a_f32(arow, 0, g), a_nxt = a_cur;
#pragma unroll
    for (int kt = 0; kt < 2; ++kt) {
      if (kt == 0) a_nxt = load_a_f32(arow, 32, g);
#pragma unroll
      for (int nt = 0; nt < 4; ++nt) {
        const int nxt = kt * 4 + nt + 2;
        v16h b2 = (nxt < 8) ? bp[nxt * 32] : b0;
        acc[nt] = wmma_f16(a_cur, b0, acc[nt]);
        b0 = b1;
        b1 = b2;
      }
      a_cur = a_nxt;
    }
  } else {
    const _Float16* arow = H2 + (size_t)(m0 - NUM_USER + n) * DL;
    v16h a_cur = load_a_f16(arow, 0, g), a_nxt = a_cur;
#pragma unroll
    for (int kt = 0; kt < 2; ++kt) {
      if (kt == 0) a_nxt = load_a_f16(arow, 32, g);
#pragma unroll
      for (int nt = 0; nt < 4; ++nt) {
        const int nxt = kt * 4 + nt + 2;
        v16h b2 = (nxt < 8) ? bp[nxt * 32] : b0;
        acc[nt] = wmma_f16(a_cur, b0, acc[nt]);
        b0 = b1;
        b1 = b2;
      }
      a_cur = a_nxt;
    }
  }

  const int mbase = m0 + (lane >> 4) * 8;
#pragma unroll
  for (int nt = 0; nt < 4; ++nt) {
    const float bz = bias[nt * 16 + n];
#pragma unroll
    for (int r = 0; r < 8; ++r)
      O[(size_t)(mbase + r) * DL + nt * 16 + n] = acc[nt][r] + bz;
  }
}

// ---------------------------------------------------------------------------
// info_nce part 1: per-group sum of L2-normalized rows  u[G][64]
// ---------------------------------------------------------------------------
__global__ __launch_bounds__(128) void infonce_u_kernel(
    const float* __restrict__ v0, const float* __restrict__ v1,
    float* __restrict__ u0, float* __restrict__ u1) {
  __shared__ float inv0[M_ROWS];
  __shared__ float inv1[M_ROWS];
  const int grp = blockIdx.x;
  const float* p0 = v0 + (size_t)grp * M_ROWS * DL;
  const float* p1 = v1 + (size_t)grp * M_ROWS * DL;
  for (int r = threadIdx.x; r < M_ROWS; r += 128) {
    float s0 = 0.f, s1 = 0.f;
    for (int d = 0; d < DL; ++d) {
      float a = p0[r * DL + d]; s0 += a * a;
      float b = p1[r * DL + d]; s1 += b * b;
    }
    inv0[r] = 1.0f / fmaxf(sqrtf(s0), 1e-12f);
    inv1[r] = 1.0f / fmaxf(sqrtf(s1), 1e-12f);
  }
  __syncthreads();
  if (threadIdx.x < 64) {               // waves 0-1: view0 (wave-uniform)
    int d = threadIdx.x;
    float s = 0.f;
    for (int r = 0; r < M_ROWS; ++r) s += p0[r * DL + d] * inv0[r];
    u0[grp * DL + d] = s;
  } else {                              // waves 2-3: view1
    int d = threadIdx.x - 64;
    float s = 0.f;
    for (int r = 0; r < M_ROWS; ++r) s += p1[r * DL + d] * inv1[r];
    u1[grp * DL + d] = s;
  }
}

// ---------------------------------------------------------------------------
// info_nce part 2: loss = (S00.sum+S01.sum-pos)/pos with
//   pos = sum_g u0_g.u0_g + u0_g.u1_g ; S00.sum = |sum_g u0_g|^2 ;
//   S01.sum = (sum u0).(sum u1)
// ---------------------------------------------------------------------------
__global__ __launch_bounds__(64) void infonce_loss_kernel(
    const float* __restrict__ u0, const float* __restrict__ u1,
    float* __restrict__ out) {
  __shared__ float sp[64];
  __shared__ float st[64];
  const int d = threadIdx.x;
  float t0 = 0.f, t1 = 0.f, a = 0.f, b = 0.f;
  for (int g = 0; g < G_GROUPS; ++g) {
    float x = u0[g * DL + d], y = u1[g * DL + d];
    t0 += x; t1 += y; a += x * x; b += x * y;
  }
  sp[d] = a + b;                 // contribution to pos
  st[d] = t0 * t0 + t0 * t1;     // contribution to S00.sum + S01.sum
  __syncthreads();
  if (d == 0) {
    float pos = 0.f, tot = 0.f;
    for (int i = 0; i < 64; ++i) { pos += sp[i]; tot += st[i]; }
    out[0] = (tot - pos) / pos;
  }
}

// ---------------------------------------------------------------------------
extern "C" void kernel_launch(void* const* d_in, const int* in_sizes, int n_in,
                              void* d_out, int out_size, void* d_ws, size_t ws_size,
                              hipStream_t stream) {
  const float* features = (const float*)d_in[0];
  const float* users    = (const float*)d_in[1];
  const float* view0    = (const float*)d_in[2];
  const float* view1    = (const float*)d_in[3];
  const float* W1 = (const float*)d_in[4];
  const float* b1 = (const float*)d_in[5];
  const float* W2 = (const float*)d_in[6];
  const float* b2 = (const float*)d_in[7];
  const float* W3 = (const float*)d_in[8];
  const float* b3 = (const float*)d_in[9];
  float* out = (float*)d_out;

  // Workspace carve-up (~26.7 MB total), 256B aligned.
  char* ws = (char*)d_ws;
  size_t off = 0;
  auto carve = [&](size_t bytes) {
    size_t o = off;
    off = (off + bytes + 255) & ~(size_t)255;
    return o;
  };
  _Float16* h1  = (_Float16*)(ws + carve((size_t)NUM_ITEM * HID * 2));
  _Float16* h2  = (_Float16*)(ws + carve((size_t)NUM_ITEM * DL  * 2));
  _Float16* w1p = (_Float16*)(ws + carve((size_t)FEAT_DIM * HID * 2));
  _Float16* w2p = (_Float16*)(ws + carve((size_t)HID * DL * 2));
  _Float16* w3p = (_Float16*)(ws + carve((size_t)DL  * DL * 2));
  float*    u0  = (float*)   (ws + carve((size_t)G_GROUPS * DL * 4));
  float*    u1  = (float*)   (ws + carve((size_t)G_GROUPS * DL * 4));
  (void)in_sizes; (void)n_in; (void)out_size; (void)ws_size;

  // 1. Pack weights into WMMA B-fragment layout (tiny, one-time per call).
  {
    int t1 = (FEAT_DIM / 32) * (HID / 16) * 32;  // 32768
    pack_w_kernel<<<(t1 + 255) / 256, 256, 0, stream>>>(W1, w1p, FEAT_DIM, HID);
    int t2 = (HID / 32) * (DL / 16) * 32;        // 1024
    pack_w_kernel<<<(t2 + 255) / 256, 256, 0, stream>>>(W2, w2p, HID, DL);
    int t3 = (DL / 32) * (DL / 16) * 32;         // 256
    pack_w_kernel<<<(t3 + 255) / 256, 256, 0, stream>>>(W3, w3p, DL, DL);
  }

  // 2. The MLP: three WMMA GEMMs.
  {
    int tiles = NUM_ITEM / 16;                   // 2500
    int blocks = (tiles + 7) / 8;                // 313 (tail waves inactive)
    gemm1_kernel<<<blocks, 256, 0, stream>>>(features, w1p, b1, h1);
    gemm2_kernel<<<blocks, 256, 0, stream>>>(h1, w2p, b2, h2);
  }
  {
    int tiles = NROWS / 16;                      // 5625
    int blocks = (tiles + 7) / 8;                // 704
    gemm3_kernel<<<blocks, 256, 0, stream>>>(users, h2, w3p, b3, out);
  }

  // 3. Contrastive loss (scalar appended after x1).
  infonce_u_kernel<<<G_GROUPS, 128, 0, stream>>>(view0, view1, u0, u1);
  infonce_loss_kernel<<<1, 64, 0, stream>>>(u0, u1, out + (size_t)NROWS * DL);
}
